// Llama4TextMoe_45251775431030
// MI455X (gfx1250) — matrix operand order repaired and checked by reading, not compiled, not run on hardware
//
#include <hip/hip_runtime.h>
#include <math.h>

#define E_EXP 16
#define H_DIM 2048
#define IDIM  1024
#define T_TOK 4096
#define BPITCH 40   // LDS row pitch (bf16 elems): 80B = 20 dwords -> conflict-free operand reads

typedef __attribute__((ext_vector_type(16))) __bf16 v16bf;
typedef __attribute__((ext_vector_type(8)))  __bf16 v8bf;
typedef __attribute__((ext_vector_type(2)))  __bf16 v2bf;
typedef __attribute__((ext_vector_type(8)))  float  v8f;

// ---------------------------------------------------------------------------
// A-operand loader: contiguous-k A 16x32 order used here, same as B (wave32):
//   lane<16 : M=lane,    elems 0..15 = K 0..15
//   lane>=16: M=lane-16, elems 0..15 = K 16..31
// ---------------------------------------------------------------------------
__device__ __forceinline__ v16bf load_a_bf16(const float* __restrict__ p, int kb, float scl) {
  v16bf a;
#pragma unroll
  for (int j = 0; j < 8; ++j) a[j] = (__bf16)(p[kb + j] * scl);
#pragma unroll
  for (int j = 0; j < 8; ++j) a[8 + j] = (__bf16)(p[kb + 8 + j] * scl);
  return a;
}

// B operand: 16 contiguous K values per lane; rows are 16B-aligned (pitch 80B),
// so assemble the 32B v16bf from two 16B v8bf LDS loads (2x ds_load_b128).
__device__ __forceinline__ v16bf load_b_lds(const __bf16* p) {
  v8bf lo = *(const v8bf*)p;
  v8bf hi = *(const v8bf*)(p + 8);
  return __builtin_shufflevector(lo, hi, 0, 1, 2, 3, 4, 5, 6, 7,
                                 8, 9, 10, 11, 12, 13, 14, 15);
}

// ---------------------------------------------------------------------------
// Router: logits = x @ router_w.T ; top-1 ; sigmoid ; scores -> d_out tail;
// per-expert token lists via atomics; per-token scale for the routed path.
// ---------------------------------------------------------------------------
__global__ __launch_bounds__(256) void router_kernel(
    const float* __restrict__ x, const float* __restrict__ rw,
    float* __restrict__ scores, int* __restrict__ counts,
    int* __restrict__ lists, float* __restrict__ tokScale)
{
  const int t = blockIdx.x;
  const int tid = threadIdx.x;
  const float* xr = x + (size_t)t * H_DIM;

  float acc[E_EXP];
#pragma unroll
  for (int e = 0; e < E_EXP; ++e) acc[e] = 0.0f;

  for (int h = tid; h < H_DIM; h += 256) {
    float xv = xr[h];
#pragma unroll
    for (int e = 0; e < E_EXP; ++e) acc[e] += xv * rw[e * H_DIM + h];
  }

#pragma unroll
  for (int off = 16; off >= 1; off >>= 1) {
#pragma unroll
    for (int e = 0; e < E_EXP; ++e) acc[e] += __shfl_xor(acc[e], off, 32);
  }

  __shared__ float red[E_EXP];
  if (tid < E_EXP) red[tid] = 0.0f;
  __syncthreads();
  if ((tid & 31) == 0) {
#pragma unroll
    for (int e = 0; e < E_EXP; ++e) atomicAdd(&red[e], acc[e]);
  }
  __syncthreads();

  if (tid == 0) {
    int am = 0; float mv = red[0];
#pragma unroll
    for (int e = 1; e < E_EXP; ++e) if (red[e] > mv) { mv = red[e]; am = e; }
    float sc = 1.0f / (1.0f + expf(-mv));
#pragma unroll
    for (int e = 0; e < E_EXP; ++e) scores[(size_t)e * T_TOK + t] = (e == am) ? sc : 0.0f;
    tokScale[t] = sc;
    int pos = atomicAdd(&counts[am], 1);
    lists[am * T_TOK + pos] = t;
  }
}

__global__ void init_counts(int* c) { if (threadIdx.x < E_EXP) c[threadIdx.x] = 0; }

// ---------------------------------------------------------------------------
// Fused gate/up GEMM + SiLU epilogue.  Block = 128 thr (4 waves), tile M=64,
// N = 64 gate + 64 up cols, K = H_DIM.  Double-buffered bf16 n-major LDS tile;
// iter i: prefetch tile i+1 (global->regs), WMMA on Bs[buf], pack regs ->
// Bs[buf^1] (v_cvt_pk_bf16_f32 + ds_store_b32), ONE barrier per iter.
// Staging index maps are affine in tid with per-it constants (hoisted bases).
// ---------------------------------------------------------------------------
template<bool GATHER, bool BNMAJOR>
__global__ __launch_bounds__(128) void gateup_kernel(
    const float* __restrict__ X,
    const float* __restrict__ WgBase, const float* __restrict__ WuBase,
    float* __restrict__ act,
    const int* __restrict__ counts, const int* __restrict__ lists,
    const float* __restrict__ tokScale, int mtPerE)
{
  const int tid  = threadIdx.x;
  const int wid  = tid >> 5;
  const int lane = tid & 31;
  const int ng   = blockIdx.y * 64;

  int mt, cnt = T_TOK;
  const int* lst = nullptr;
  const float *Wg, *Wu;
  int ldb;
  if constexpr (GATHER) {
    int e = blockIdx.x / mtPerE;
    mt = blockIdx.x % mtPerE;
    cnt = counts[e];
    if (mt * 64 >= cnt) return;              // uniform across block
    lst = lists + e * T_TOK;
    Wg  = WgBase + (size_t)e * H_DIM * (2 * IDIM);
    Wu  = Wg + IDIM;                         // up = same array, +I column offset
    ldb = 2 * IDIM;
  } else {
    mt = blockIdx.x; Wg = WgBase; Wu = WuBase; ldb = H_DIM;
  }

  // per-lane A row (loop invariant)
  const int mrow = mt * 64 + wid * 16 + (lane & 15);
  const float* arow;
  float scl;
  if constexpr (GATHER) {
    bool valid = mrow < cnt;
    int tok = valid ? lst[mrow] : 0;
    arow = X + (size_t)tok * H_DIM;
    scl  = valid ? tokScale[tok] : 0.0f;     // invalid rows compute zeros
  } else {
    arow = X + (size_t)mrow * H_DIM;
    scl  = 1.0f;
  }
  const int kb  = (lane >> 4) * 16;
  const int kbB = (lane >> 4) * 16;

  __shared__ alignas(64) __bf16 Bs[2][128][BPITCH];  // n-major, double buffered

  // B tile staging: 128(N) x 32(K), 16 bf16-pairs per thread.
  float vr0[16], vr1[16];
  auto loadB = [&](int k0) {
    if constexpr (BNMAJOR) {
      const int rr = tid >> 4;               // 0..7
      const int kk = (tid & 15) * 2;         // even k within tile
#pragma unroll
      for (int it = 0; it < 16; ++it) {
        int nl = (it & 7) * 8 + rr;          // row within 64-col half
        const float* src = (it < 8) ? Wg : Wu;
        const float* p = src + (size_t)(ng + nl) * ldb + (k0 + kk);
        vr0[it] = p[0]; vr1[it] = p[1];      // contiguous pair -> b64 load
      }
    } else {
      const float* p0 = ((tid < 64) ? Wg : Wu) + (ng + (tid & 63));
#pragma unroll
      for (int it = 0; it < 16; ++it) {
        vr0[it] = p0[(size_t)(k0 + 2 * it) * ldb];
        vr1[it] = p0[(size_t)(k0 + 2 * it + 1) * ldb];
      }
    }
  };
  auto storeB = [&](int buf) {
    __bf16* base = &Bs[buf][0][0];
    const int sBase = BNMAJOR ? ((tid >> 4) * BPITCH + (tid & 15) * 2)
                              : (tid * BPITCH);
#pragma unroll
    for (int it = 0; it < 16; ++it) {
      int off = BNMAJOR ? (sBase + it * 8 * BPITCH)   // nl = it*8 + tid>>4
                        : (sBase + it * 2);           // kp = 2*it
      v2bf pk; pk[0] = (__bf16)vr0[it]; pk[1] = (__bf16)vr1[it];
      *(v2bf*)(base + off) = pk;                      // packed cvt + ds_store_b32
    }
  };

  v8f accG[4] = {}; v8f accU[4] = {};

  loadB(0);
  storeB(0);
  __syncthreads();

  int buf = 0;
  for (int k0 = 0; k0 < H_DIM; k0 += 32, buf ^= 1) {
    const bool hasNext = (k0 + 32) < H_DIM;
    if (hasNext) loadB(k0 + 32);              // prefetch next tile (global)

    v16bf a = load_a_bf16(arow + k0, kb, scl);
#pragma unroll
    for (int ns = 0; ns < 4; ++ns) {
      v16bf bg = load_b_lds(&Bs[buf][ns * 16 + (lane & 15)][kbB]);
      accG[ns] = __builtin_amdgcn_wmma_f32_16x16x32_bf16(
          false, a, false, bg, (short)0, accG[ns], false, false);
      v16bf bu = load_b_lds(&Bs[buf][64 + ns * 16 + (lane & 15)][kbB]);
      accU[ns] = __builtin_amdgcn_wmma_f32_16x16x32_bf16(
          false, a, false, bu, (short)0, accU[ns], false, false);
    }

    if (hasNext) storeB(buf ^ 1);             // fill other buffer
    __syncthreads();                          // single barrier per k-step
  }

  // epilogue: act = up * silu(gate), fp32 -> ws
#pragma unroll
  for (int r = 0; r < 8; ++r) {
    int mloc = wid * 16 + ((lane < 16) ? r : r + 8);   // C/D layout: lane>=16 -> M+8
    int grow = mt * 64 + mloc;
    int tok;
    if constexpr (GATHER) { if (grow >= cnt) continue; tok = lst[grow]; }
    else tok = grow;
    float* orow = act + (size_t)tok * IDIM + ng + (lane & 15);
#pragma unroll
    for (int ns = 0; ns < 4; ++ns) {
      float g = accG[ns][r], u = accU[ns][r];
      orow[ns * 16] = u * g / (1.0f + expf(-g));       // u * silu(g)
    }
  }
}

// ---------------------------------------------------------------------------
// Down GEMM: act[T,I] (fp32 ws) @ down-weights -> out[T,H].
// Same double-buffered pipeline, 64-col tile, K = IDIM.
// ACCUM=true adds (routed contribution after shared wrote), else stores.
// ---------------------------------------------------------------------------
template<bool GATHER, bool BNMAJOR, bool ACCUM>
__global__ __launch_bounds__(128) void down_kernel(
    const float* __restrict__ Aact, const float* __restrict__ WBase,
    float* __restrict__ outp,
    const int* __restrict__ counts, const int* __restrict__ lists, int mtPerE)
{
  const int tid  = threadIdx.x;
  const int wid  = tid >> 5;
  const int lane = tid & 31;
  const int ng   = blockIdx.y * 64;

  int mt, cnt = T_TOK;
  const int* lst = nullptr;
  const float* W;
  int ldb;
  if constexpr (GATHER) {
    int e = blockIdx.x / mtPerE;
    mt = blockIdx.x % mtPerE;
    cnt = counts[e];
    if (mt * 64 >= cnt) return;
    lst = lists + e * T_TOK;
    W   = WBase + (size_t)e * IDIM * H_DIM;  // down_proj[e]: (k,n) at k*H + n
    ldb = H_DIM;
  } else {
    mt = blockIdx.x; W = WBase; ldb = IDIM;  // down_w: (k,n) at n*I + k
  }

  const int mrow = mt * 64 + wid * 16 + (lane & 15);
  const float* arow;
  float scl = 1.0f;
  if constexpr (GATHER) {
    bool valid = mrow < cnt;
    int tok = valid ? lst[mrow] : 0;
    arow = Aact + (size_t)tok * IDIM;
    if (!valid) scl = 0.0f;
  } else {
    arow = Aact + (size_t)mrow * IDIM;
  }
  const int kb  = (lane >> 4) * 16;
  const int kbB = (lane >> 4) * 16;

  __shared__ alignas(64) __bf16 Bs[2][64][BPITCH];

  // 64(N) x 32(K), 8 bf16-pairs per thread
  float vr0[8], vr1[8];
  auto loadB = [&](int k0) {
    if constexpr (BNMAJOR) {
      const int rr = tid >> 4;
      const int kk = (tid & 15) * 2;
#pragma unroll
      for (int it = 0; it < 8; ++it) {
        const float* p = W + (size_t)(ng + it * 8 + rr) * ldb + (k0 + kk);
        vr0[it] = p[0]; vr1[it] = p[1];
      }
    } else {
      const float* p0 = W + (ng + (tid & 63));
      const int kk = (tid >> 6) * 2;          // 0 or 2
#pragma unroll
      for (int it = 0; it < 8; ++it) {
        vr0[it] = p0[(size_t)(k0 + it * 4 + kk) * ldb];
        vr1[it] = p0[(size_t)(k0 + it * 4 + kk + 1) * ldb];
      }
    }
  };
  auto storeB = [&](int buf) {
    __bf16* base = &Bs[buf][0][0];
    const int sBase = BNMAJOR ? ((tid >> 4) * BPITCH + (tid & 15) * 2)
                              : ((tid & 63) * BPITCH + (tid >> 6) * 2);
#pragma unroll
    for (int it = 0; it < 8; ++it) {
      int off = BNMAJOR ? (sBase + it * 8 * BPITCH)   // nl = it*8 + tid>>4
                        : (sBase + it * 4);           // kp = it*4 + (tid>>6)*2
      v2bf pk; pk[0] = (__bf16)vr0[it]; pk[1] = (__bf16)vr1[it];
      *(v2bf*)(base + off) = pk;
    }
  };

  v8f acc[4] = {};

  loadB(0);
  storeB(0);
  __syncthreads();

  int buf = 0;
  for (int k0 = 0; k0 < IDIM; k0 += 32, buf ^= 1) {
    const bool hasNext = (k0 + 32) < IDIM;
    if (hasNext) loadB(k0 + 32);

    v16bf a = load_a_bf16(arow + k0, kb, scl);
#pragma unroll
    for (int ns = 0; ns < 4; ++ns) {
      v16bf b = load_b_lds(&Bs[buf][ns * 16 + (lane & 15)][kbB]);
      acc[ns] = __builtin_amdgcn_wmma_f32_16x16x32_bf16(
          false, a, false, b, (short)0, acc[ns], false, false);
    }

    if (hasNext) storeB(buf ^ 1);
    __syncthreads();
  }

#pragma unroll
  for (int r = 0; r < 8; ++r) {
    int mloc = wid * 16 + ((lane < 16) ? r : r + 8);
    int grow = mt * 64 + mloc;
    int tok;
    if constexpr (GATHER) { if (grow >= cnt) continue; tok = lst[grow]; }
    else tok = grow;
    float* orow = outp + (size_t)tok * H_DIM + ng + (lane & 15);
#pragma unroll
    for (int ns = 0; ns < 4; ++ns) {
      if constexpr (ACCUM) orow[ns * 16] += acc[ns][r];
      else                 orow[ns * 16]  = acc[ns][r];
    }
  }
}

// ---------------------------------------------------------------------------
extern "C" void kernel_launch(void* const* d_in, const int* in_sizes, int n_in,
                              void* d_out, int out_size, void* d_ws, size_t ws_size,
                              hipStream_t stream)
{
  (void)in_sizes; (void)n_in; (void)out_size; (void)ws_size;
  const float* x   = (const float*)d_in[0];  // [B,S,H] -> [T,H]
  const float* rw  = (const float*)d_in[1];  // [E,H]
  const float* gup = (const float*)d_in[2];  // [E,H,2I]
  const float* dwn = (const float*)d_in[3];  // [E,I,H]
  const float* gw  = (const float*)d_in[4];  // [I,H]
  const float* uw  = (const float*)d_in[5];  // [I,H]
  const float* dw  = (const float*)d_in[6];  // [H,I]

  float* out    = (float*)d_out;                       // [T,H]
  float* scores = out + (size_t)T_TOK * H_DIM;         // [E,T]

  char* ws = (char*)d_ws;
  int*   counts   = (int*)ws;    ws += 256;
  int*   lists    = (int*)ws;    ws += (size_t)E_EXP * T_TOK * sizeof(int);
  float* tokScale = (float*)ws;  ws += (size_t)T_TOK * sizeof(float);
  float* act_r    = (float*)ws;  ws += (size_t)T_TOK * IDIM * sizeof(float);
  float* act_s    = (float*)ws;  // total ws use ~32.3 MB

  const int mtPerE = T_TOK / 64;

  init_counts<<<1, 32, 0, stream>>>(counts);
  router_kernel<<<T_TOK, 256, 0, stream>>>(x, rw, scores, counts, lists, tokScale);

  // shared expert: act_s = silu(x@gate_w.T) * (x@up_w.T); out = act_s @ down_w.T
  gateup_kernel<false, true><<<dim3(T_TOK / 64, IDIM / 64), 128, 0, stream>>>(
      x, gw, uw, act_s, nullptr, nullptr, nullptr, 0);
  down_kernel<false, true, false><<<dim3(T_TOK / 64, H_DIM / 64), 128, 0, stream>>>(
      act_s, dw, out, nullptr, nullptr, 0);

  // routed top-1 experts (grouped GEMM over gathered token lists)
  gateup_kernel<true, false><<<dim3(E_EXP * mtPerE, IDIM / 64), 128, 0, stream>>>(
      x, gup, nullptr, act_r, counts, lists, tokScale, mtPerE);
  down_kernel<true, false, true><<<dim3(E_EXP * mtPerE, H_DIM / 64), 128, 0, stream>>>(
      act_r, dwn, out, counts, lists, mtPerE);
}
